// Model_76974403879581
// MI455X (gfx1250) — matrix operand order repaired and checked by reading, not compiled, not run on hardware
//
#include <hip/hip_runtime.h>

// ---------------------------------------------------------------------------
// Grouped block-scaled GEMM-NT for MI455X (gfx1250, wave32, WMMA).
//   D[m, n] = sum_k (a[m,k] * a_sf[m, k/128]) * (b[g, n, k] * b_sf[g, n/128, k/128])
//   with g = m_indices[m] (sorted), output bf16.
// Scales folded into the fp32->bf16 conversion during LDS staging; WMMA f32
// accumulators chain across all of K. Double-buffered LDS + explicit software
// pipeline: global loads for chunk k+1 are in flight while WMMAs run chunk k.
// Per k-step, all 12 fragment ds_loads issue before the 8 WMMAs so the
// scheduler can use partial s_wait_dscnt and overlap LDS latency with XDL.
// ---------------------------------------------------------------------------

typedef __attribute__((ext_vector_type(16))) __bf16 v16bf;
typedef __attribute__((ext_vector_type(8)))  float  v8f;

#define M_DIM  4096
#define N_DIM  2048
#define K_DIM  2048
#define BLK    128
#define KB     (K_DIM / BLK)     // 16 scale blocks along K
#define BM     128               // M tile per workgroup
#define BN     128               // N tile per workgroup (== BLK: one b_sf per tile/kb)
#define BKC    64                // K chunk staged in LDS per pipeline stage
#define NCHUNK (K_DIM / BKC)     // 32
#define LDT    72                // padded LDS row stride in bf16 elems (144B, 16B
                                 // aligned, 9 x 4-dword step -> conflict-free b128)

union Frag { uint4 u[2]; v16bf v; };

// round-to-nearest-even pack of two floats into two bf16 in one dword
__device__ __forceinline__ unsigned pack2_bf16(float a, float b) {
  union { float f; unsigned u; } x, y;
  x.f = a; y.f = b;
  unsigned lo = (x.u + 0x7FFFu + ((x.u >> 16) & 1u)) >> 16;
  unsigned hi = (y.u + 0x7FFFu + ((y.u >> 16) & 1u)) >> 16;
  return (lo & 0xFFFFu) | (hi << 16);
}

__global__ __launch_bounds__(256)
void grouped_scaled_gemm_bf16(const float* __restrict__ a_data,
                              const float* __restrict__ a_sf,
                              const float* __restrict__ b_data,
                              const float* __restrict__ b_sf,
                              const int*   __restrict__ m_indices,
                              unsigned short* __restrict__ out /* bf16 bits */) {
  __shared__ unsigned short As[2][BM * LDT];  // bf16 bits, scales pre-applied
  __shared__ unsigned short Bs[2][BN * LDT];

  const int t     = threadIdx.x;
  const int lane  = t & 31;
  const int wid   = t >> 5;              // 0..7
  const int waveM = (wid >> 2) * 64;     // 0 or 64
  const int waveN = (wid & 3) * 32;      // 0,32,64,96
  const int lh    = lane >> 4;           // lane half
  const int l16   = lane & 15;

  // staging slot: fixed 4-bf16 column group per thread, 8 rows (stride 16)
  const int sr = t >> 4;                 // base row 0..15
  const int sc = (t & 15) * 4;           // col offset (floats == bf16 elems)

  const int m0 = blockIdx.y * BM;
  const int n0 = blockIdx.x * BN;
  const int nb = blockIdx.x;             // BN == BLK

  // Sorted m_indices: this tile's rows use groups in [g_lo, g_hi].
  const int g_lo = m_indices[m0];
  const int g_hi = m_indices[m0 + BM - 1];
  const int gs   = g_lo < 0 ? 0 : g_lo;
  const int ge   = g_hi < gs ? gs : g_hi;

  for (int g = gs; g <= ge; ++g) {
    v8f acc[4][2];
#pragma unroll
    for (int mi = 0; mi < 4; ++mi)
#pragma unroll
      for (int ni = 0; ni < 2; ++ni)
        acc[mi][ni] = (v8f){0.f, 0.f, 0.f, 0.f, 0.f, 0.f, 0.f, 0.f};

    // ---- prologue: stage chunk 0 into buffer 0 ----
    {
      const float bsc0 = b_sf[((size_t)g * (N_DIM / BLK) + nb) * KB];
#pragma unroll
      for (int it = 0; it < 8; ++it) {
        const int row = sr + it * 16;
        const float4 av = *(const float4*)(a_data + (size_t)(m0 + row) * K_DIM + sc);
        const float  sa = a_sf[(size_t)(m0 + row) * KB];
        uint2 pa;
        pa.x = pack2_bf16(av.x * sa, av.y * sa);
        pa.y = pack2_bf16(av.z * sa, av.w * sa);
        *(uint2*)(As[0] + row * LDT + sc) = pa;
        const float4 bv = *(const float4*)(b_data + ((size_t)g * N_DIM + n0 + row) * K_DIM + sc);
        uint2 pb;
        pb.x = pack2_bf16(bv.x * bsc0, bv.y * bsc0);
        pb.y = pack2_bf16(bv.z * bsc0, bv.w * bsc0);
        *(uint2*)(Bs[0] + row * LDT + sc) = pb;
      }
    }

    int cur = 0;
    for (int kc = 0; kc < NCHUNK; ++kc) {
      __syncthreads();  // staged buffer 'cur' visible; prior reads of 'cur^1' done

      // ---- phase 1: issue global loads for chunk kc+1 (no waits here) ----
      float4 aval[8], bval[8];
      float  asc[8], bsc = 0.f;
      const bool more = (kc + 1 < NCHUNK);
      if (more) {
        const int kg = (kc + 1) * BKC;
        const int kb = (kc + 1) >> 1;
#pragma unroll
        for (int it = 0; it < 8; ++it) {
          const int row = sr + it * 16;
          aval[it] = *(const float4*)(a_data + (size_t)(m0 + row) * K_DIM + kg + sc);
          asc[it]  = a_sf[(size_t)(m0 + row) * KB + kb];
          bval[it] = *(const float4*)(b_data + ((size_t)g * N_DIM + n0 + row) * K_DIM + kg + sc);
        }
        bsc = b_sf[((size_t)g * (N_DIM / BLK) + nb) * KB + kb];
      }

      // ---- compute chunk kc from LDS buffer 'cur' ----
      const unsigned short* curA = As[cur];
      const unsigned short* curB = Bs[cur];
#pragma unroll
      for (int ks = 0; ks < 2; ++ks) {
        const int k0 = ks * 32;
        // Issue ALL fragment loads first, then the 8 WMMAs: lets the
        // scheduler overlap ds_load latency with XDL via partial dscnt waits.
        // B 32x16 bf16 fragment: lane holds column n=l16; lane<16: K0..15,
        // lane>=16: K16..31 (contiguous along K, matching N-major B in memory)
        Frag bfr[2];
#pragma unroll
        for (int ni = 0; ni < 2; ++ni) {
          const unsigned short* p = curB + (waveN + ni * 16 + l16) * LDT + k0 + lh * 16;
          bfr[ni].u[0] = *(const uint4*)(p);
          bfr[ni].u[1] = *(const uint4*)(p + 8);
        }
        // A 16x32 bf16 fragment: lane<16: K0..15; lane>=16: K16..31
        // (contiguous along K, matching B).
        Frag afr[4];
#pragma unroll
        for (int mi = 0; mi < 4; ++mi) {
          const unsigned short* p = curA + (waveM + mi * 16 + l16) * LDT + k0;
          afr[mi].u[0] = *(const uint4*)(p + lh * 16);
          afr[mi].u[1] = *(const uint4*)(p + lh * 16 + 8);
        }
#pragma unroll
        for (int mi = 0; mi < 4; ++mi)
#pragma unroll
          for (int ni = 0; ni < 2; ++ni)
            acc[mi][ni] = __builtin_amdgcn_wmma_f32_16x16x32_bf16(
                false, afr[mi].v, false, bfr[ni].v,
                (short)0, acc[mi][ni], false, false);
      }

      // ---- phase 2: scale-convert and store chunk kc+1 into buffer cur^1 ----
      if (more) {
        unsigned short* nA = As[cur ^ 1];
        unsigned short* nB = Bs[cur ^ 1];
#pragma unroll
        for (int it = 0; it < 8; ++it) {
          const int row = sr + it * 16;
          const float sa = asc[it];
          uint2 pa;
          pa.x = pack2_bf16(aval[it].x * sa, aval[it].y * sa);
          pa.y = pack2_bf16(aval[it].z * sa, aval[it].w * sa);
          *(uint2*)(nA + row * LDT + sc) = pa;
          uint2 pb;
          pb.x = pack2_bf16(bval[it].x * bsc, bval[it].y * bsc);
          pb.y = pack2_bf16(bval[it].z * bsc, bval[it].w * bsc);
          *(uint2*)(nB + row * LDT + sc) = pb;
        }
      }
      cur ^= 1;
    }

    // ---- masked bf16 store: row m belongs to exactly one g in [gs, ge] ----
    // C/D layout: VGPR r, lane<16 -> M=r, N=lane ; lane>=16 -> M=8+r, N=lane-16
#pragma unroll
    for (int mi = 0; mi < 4; ++mi) {
#pragma unroll
      for (int r = 0; r < 8; ++r) {
        const int mrow = m0 + waveM + mi * 16 + (lh << 3) + r;
        const int gi   = m_indices[mrow];
#pragma unroll
        for (int ni = 0; ni < 2; ++ni) {
          const int ncol = n0 + waveN + ni * 16 + l16;
          if (gi == g) {
            out[(size_t)mrow * N_DIM + ncol] =
                (unsigned short)pack2_bf16(acc[mi][ni][r], 0.f);
          } else if (gi < 0 && g == gs) {
            out[(size_t)mrow * N_DIM + ncol] = 0;  // padding rows -> 0
          }
        }
      }
    }
  }
}

extern "C" void kernel_launch(void* const* d_in, const int* in_sizes, int n_in,
                              void* d_out, int out_size, void* d_ws, size_t ws_size,
                              hipStream_t stream) {
  const float* a_data = (const float*)d_in[0];
  const float* a_sf   = (const float*)d_in[1];
  const float* b_data = (const float*)d_in[2];
  const float* b_sf   = (const float*)d_in[3];
  const int*   m_idx  = (const int*)d_in[4];
  unsigned short* outp = (unsigned short*)d_out;  // bf16 bit pattern

  dim3 grid(N_DIM / BN, M_DIM / BM);              // 16 x 32 workgroups
  grouped_scaled_gemm_bf16<<<grid, 256, 0, stream>>>(a_data, a_sf, b_data, b_sf,
                                                     m_idx, outp);
}
